// GNN_GRU_Model_49031346651548
// MI455X (gfx1250) — compile-verified, matrix operand order checked by reading, not run on hardware
//
#include <hip/hip_runtime.h>
#include <hip/hip_bf16.h>
#include <cstdint>
#include <cstddef>

#define T_STEPS 8
#define NN      20000
#define EE      320000
#define ETOT    (EE + NN)
#define FIN     32
#define HIDC    64
#define HEADS   4
#define GH      128

// GRU weight image geometry: 24 col-tiles x 4 k-chunks x 32 lanes x 16 halves
#define GRU_IMG_HALVES (24*4*32*16)
#define GRU_LDS_BYTES  (2*GRU_IMG_HALVES*2)

typedef __attribute__((ext_vector_type(16))) _Float16 v16h;
typedef __attribute__((ext_vector_type(8)))  _Float16 v8h;
typedef __attribute__((ext_vector_type(8)))  float    v8f;

#define WMMA_F16(a,b,c) __builtin_amdgcn_wmma_f32_16x16x32_f16(false,(a),false,(b),(short)0,(c),false,false)

__device__ __forceinline__ float sigf(float x){ return 1.0f/(1.0f + expf(-x)); }

// ---- WMMA operand loaders -------------------------------------------------
// 16-bit A matrix 16x32 (MxK): lanes 0-15 rows M=0..15 with K 0-7 / 16-23,
// lanes 16-31 same rows with K 8-15 / 24-31 (ISA 7.12.2 layout).
__device__ __forceinline__ v16h load_A16(const _Float16* __restrict__ base, int stride,
                                         int row0, int kbase, int lane){
  int r  = lane & 15;
  int hi = lane >> 4;
  const _Float16* p = base + (size_t)(row0 + r)*stride + kbase + hi*8;
  v8h lo = *(const v8h*)p;          // K = kbase + hi*8 + [0..7]
  v8h hv = *(const v8h*)(p + 16);   // K = kbase + 16 + hi*8 + [0..7]
  v16h o;
  #pragma unroll
  for (int i=0;i<8;i++){ o[i]=lo[i]; o[i+8]=hv[i]; }
  return o;
}

// B operand from a pre-packed global "register image": [ctile][kchunk][lane][16]
__device__ __forceinline__ v16h load_B16(const _Float16* __restrict__ img,
                                         int ct, int kc, int kch, int lane){
  const _Float16* p = img + ((size_t)(ct*kch + kc)*32 + lane)*16;
  v8h lo = *(const v8h*)p;
  v8h hv = *(const v8h*)(p + 8);
  v16h o;
  #pragma unroll
  for (int i=0;i<8;i++){ o[i]=lo[i]; o[i+8]=hv[i]; }
  return o;
}

// Same layout, but sourced from LDS (ds_load_b128 path)
__device__ __forceinline__ v16h lds_B16(const _Float16* base, int ct, int kc, int lane){
  const _Float16* p = base + ((size_t)(ct*4 + kc)*32 + lane)*16;
  v8h lo = *(const v8h*)p;
  v8h hv = *(const v8h*)(p + 8);
  v16h o;
  #pragma unroll
  for (int i=0;i<8;i++){ o[i]=lo[i]; o[i+8]=hv[i]; }
  return o;
}

// ---- weight image packing -------------------------------------------------
// transposed==0: B[k][n] = W[k*Ncols + n]     (x @ W)
// transposed==1: B[k][n] = W[n*Kdim + k]      (x @ W.T, W row-major [Ncols][Kdim])
__global__ void prep_bimg(const float* __restrict__ W, _Float16* __restrict__ img,
                          int Kdim, int Ncols, int transposed){
  int idx = blockIdx.x*blockDim.x + threadIdx.x;
  int ctiles = Ncols >> 4, kch = Kdim >> 5;
  int total = ctiles*kch*32;
  if (idx >= total) return;
  int lane = idx & 31;
  int kc   = (idx >> 5) % kch;
  int ct   = (idx >> 5) / kch;
  int col  = ct*16 + (lane & 15);
  int hi   = lane >> 4;
  _Float16* out = img + ((size_t)(ct*kch + kc)*32 + lane)*16;
  #pragma unroll
  for (int e=0;e<16;e++){
    int k = kc*32 + (e < 8 ? hi*8 + e : 16 + hi*8 + (e - 8));
    float v = transposed ? W[(size_t)col*Kdim + k] : W[(size_t)k*Ncols + col];
    out[e] = (_Float16)v;
  }
}

// ---- generic WMMA linear: Out[16*tiles, ncols] = A(f16) @ Bimg ------------
template<int KCH>
__global__ void wmma_linear(const _Float16* __restrict__ Ain, int astride,
                            const _Float16* __restrict__ Bimg,
                            float* __restrict__ Out, int ncols, int ctiles, int tiles){
  int wave = threadIdx.x >> 5, lane = threadIdx.x & 31;
  int tile = blockIdx.x*(blockDim.x >> 5) + wave;
  if (tile >= tiles) return;                    // wave-uniform: EXEC stays full
  int row0 = tile*16;
  v16h a[KCH];
  #pragma unroll
  for (int kc=0;kc<KCH;kc++) a[kc] = load_A16(Ain, astride, row0, kc*32, lane);
  int col0 = lane & 15, rb = (lane >> 4) << 3;
  for (int ct=0; ct<ctiles; ct++){
    v8f acc = {};
    #pragma unroll
    for (int kc=0;kc<KCH;kc++)
      acc = WMMA_F16(a[kc], load_B16(Bimg, ct, kc, KCH, lane), acc);
    int col = ct*16 + col0;
    #pragma unroll
    for (int v=0;v<8;v++)
      Out[(size_t)(row0 + rb + v)*ncols + col] = acc[v];
  }
}

// ---- fused GRU step -------------------------------------------------------
// One wave = 16 rows. Weight images staged in LDS once per block; grid-stride
// over row tiles. r/z pre-activations accumulate input+hidden WMMAs into a
// single v8f each (ir+hr, iz+hz); only the n gate keeps separate accumulators.
// LDS (192KB) pins occupancy to 1 block/WGP (2 waves/SIMD), so let the
// register allocator use a big VGPR budget instead of spilling the A operands.
__global__ void
__launch_bounds__(256, 1)
__attribute__((amdgpu_waves_per_eu(1, 2)))
gru_step(const _Float16* __restrict__ Xin,
         _Float16* __restrict__ H16, float* __restrict__ H32,
         const _Float16* __restrict__ imgWih,
         const _Float16* __restrict__ imgWhh,
         const float* __restrict__ bih, const float* __restrict__ bhh,
         int tiles){
  extern __shared__ _Float16 lds_pool[];
  _Float16* ldsIh = lds_pool;
  _Float16* ldsHh = lds_pool + GRU_IMG_HALVES;

  // cooperative stage of both weight images into LDS (v8h granularity)
  {
    const v8h* gih = (const v8h*)imgWih;
    const v8h* ghh = (const v8h*)imgWhh;
    v8h* lih = (v8h*)ldsIh;
    v8h* lhh = (v8h*)ldsHh;
    for (int i = threadIdx.x; i < GRU_IMG_HALVES/8; i += blockDim.x){
      lih[i] = gih[i];
      lhh[i] = ghh[i];
    }
  }
  __syncthreads();

  int wave = threadIdx.x >> 5, lane = threadIdx.x & 31;
  int wpb  = blockDim.x >> 5;
  int col  = lane & 15, rb = (lane >> 4) << 3;

  for (int tile = blockIdx.x*wpb + wave; tile < tiles; tile += gridDim.x*wpb){
    int row0 = tile*16;
    v16h ax[4], ah[4];
    #pragma unroll
    for (int kc=0;kc<4;kc++){
      ax[kc] = load_A16(Xin, GH, row0, kc*32, lane);
      ah[kc] = load_A16(H16, GH, row0, kc*32, lane);
    }
    for (int c=0;c<8;c++){
      v8f ar={}, az={}, ni={}, nh={};
      // interleave accumulator targets so back-to-back WMMAs are independent
      #pragma unroll
      for (int kc=0;kc<4;kc++){
        ar = WMMA_F16(ax[kc], lds_B16(ldsIh,      c, kc, lane), ar);
        az = WMMA_F16(ax[kc], lds_B16(ldsIh,  8 + c, kc, lane), az);
        ni = WMMA_F16(ax[kc], lds_B16(ldsIh, 16 + c, kc, lane), ni);
        nh = WMMA_F16(ah[kc], lds_B16(ldsHh, 16 + c, kc, lane), nh);
        ar = WMMA_F16(ah[kc], lds_B16(ldsHh,      c, kc, lane), ar);
        az = WMMA_F16(ah[kc], lds_B16(ldsHh,  8 + c, kc, lane), az);
      }
      int cg = c*16 + col;
      float br = bih[cg]      + bhh[cg];
      float bz = bih[GH + cg] + bhh[GH + cg];
      float bni = bih[2*GH + cg], bnh = bhh[2*GH + cg];
      #pragma unroll
      for (int v=0;v<8;v++){
        int row = row0 + rb + v;
        float r  = sigf(ar[v] + br);
        float z  = sigf(az[v] + bz);
        float ng = tanhf(ni[v] + bni + r*(nh[v] + bnh));
        size_t idx = (size_t)row*GH + cg;
        float hp = H32[idx];
        float hn = (1.0f - z)*ng + z*hp;
        H32[idx] = hn;
        H16[idx] = (_Float16)hn;
      }
    }
  }
}

// ---- small helper kernels -------------------------------------------------
__global__ void build_edges(const int* __restrict__ eidx, int* __restrict__ src,
                            int* __restrict__ dst){
  int i = blockIdx.x*blockDim.x + threadIdx.x;
  if (i >= ETOT) return;
  if (i < EE){ src[i] = eidx[i]; dst[i] = eidx[EE + i]; }
  else       { src[i] = i - EE;  dst[i] = i - EE; }
}

__global__ void conv_f16(const float* __restrict__ in, _Float16* __restrict__ out, int n){
  int i = blockIdx.x*blockDim.x + threadIdx.x;
  if (i < n) out[i] = (_Float16)in[i];
}

__global__ void compute_al(const float* __restrict__ h, const float* __restrict__ a_s,
                           const float* __restrict__ a_d, float* __restrict__ alS,
                           float* __restrict__ alD, int n, int H, int C){
  int i = blockIdx.x*blockDim.x + threadIdx.x;
  if (i >= n*H) return;
  int node = i / H, hd = i % H;
  const float* hp = h + (size_t)node*H*C + hd*C;
  float s = 0.f, d = 0.f;
  for (int c=0;c<C;c++){ s += hp[c]*a_s[hd*C + c]; d += hp[c]*a_d[hd*C + c]; }
  alS[i] = s; alD[i] = d;
}

__global__ void init_ms(float* __restrict__ m, float* __restrict__ s, int n){
  int i = blockIdx.x*blockDim.x + threadIdx.x;
  if (i < n){ m[i] = -1e30f; s[i] = 0.f; }
}

__device__ __forceinline__ void atomicMaxF(float* addr, float v){
  int old = __float_as_int(*addr);
  while (v > __int_as_float(old)){
    int prev = atomicCAS((int*)addr, old, __float_as_int(v));
    if (prev == old) break;
    old = prev;
  }
}

__global__ void attn_edge1(const int* __restrict__ src, const int* __restrict__ dst,
                           const float* __restrict__ alS, const float* __restrict__ alD,
                           float* __restrict__ ebuf, float* __restrict__ m, int H){
  int i = blockIdx.x*blockDim.x + threadIdx.x;
  if (i >= ETOT*H) return;
  int e = i / H, hd = i % H;
  float v = alS[src[e]*H + hd] + alD[dst[e]*H + hd];
  v = v > 0.f ? v : 0.2f*v;                  // leaky_relu(0.2)
  ebuf[i] = v;
  atomicMaxF(&m[dst[e]*H + hd], v);
}

__global__ void attn_edge2(const int* __restrict__ dst, float* __restrict__ ebuf,
                           const float* __restrict__ m, float* __restrict__ s, int H){
  int i = blockIdx.x*blockDim.x + threadIdx.x;
  if (i >= ETOT*H) return;
  int e = i / H, hd = i % H;
  float ex = expf(ebuf[i] - m[dst[e]*H + hd]);
  ebuf[i] = ex;
  atomicAdd(&s[dst[e]*H + hd], ex);
}

__global__ void attn_edge3(const int* __restrict__ src, const int* __restrict__ dst,
                           const float* __restrict__ ebuf, const float* __restrict__ s,
                           const float* __restrict__ hsrc, float* __restrict__ out,
                           int H, int C){
  int i = blockIdx.x*blockDim.x + threadIdx.x;
  if (i >= ETOT*H) return;
  int e = i / H, hd = i % H;
  float att = ebuf[i] / (s[dst[e]*H + hd] + 1e-16f);
  const float* hp = hsrc + (size_t)src[e]*H*C + hd*C;
  float*       op = out  + (size_t)dst[e]*H*C + hd*C;
  for (int c=0;c<C;c++) atomicAdd(&op[c], hp[c]*att);
}

__global__ void finalize_elu(const float* __restrict__ in, const float* __restrict__ b,
                             _Float16* __restrict__ out, int n, int cols){
  int i = blockIdx.x*blockDim.x + threadIdx.x;
  if (i >= n) return;
  float v = in[i] + b[i % cols];
  v = v > 0.f ? v : (expf(v) - 1.0f);        // elu
  out[i] = (_Float16)v;
}

__global__ void finalize_add(const float* __restrict__ in, const float* __restrict__ b,
                             _Float16* __restrict__ out, int n, int cols){
  int i = blockIdx.x*blockDim.x + threadIdx.x;
  if (i >= n) return;
  out[i] = (_Float16)(in[i] + b[i % cols]);
}

__global__ void gather_emb(const int* __restrict__ eidx, const _Float16* __restrict__ h2,
                           _Float16* __restrict__ emb){
  int i = blockIdx.x*blockDim.x + threadIdx.x;
  if (i >= EE*HIDC) return;
  int e = i / HIDC, c = i % HIDC;
  emb[(size_t)e*GH + c]        = h2[(size_t)eidx[e]*HIDC + c];       // src half
  emb[(size_t)e*GH + HIDC + c] = h2[(size_t)eidx[EE + e]*HIDC + c];  // dst half
}

__global__ void decoder_k(const float* __restrict__ H, const float* __restrict__ Wd1,
                          const float* __restrict__ bd1, const float* __restrict__ Wd2,
                          const float* __restrict__ bd2, float* __restrict__ out, int n){
  int i = blockIdx.x*blockDim.x + threadIdx.x;
  if (i >= n) return;
  const float* h = H + (size_t)i*GH;
  float acc[64];
  #pragma unroll
  for (int j=0;j<64;j++) acc[j] = bd1[j];
  for (int k=0;k<GH;k++){
    float hv = h[k];
    const float* w = Wd1 + k*64;
    #pragma unroll
    for (int j=0;j<64;j++) acc[j] += hv*w[j];
  }
  float o = bd2[0];
  #pragma unroll
  for (int j=0;j<64;j++){
    float a = acc[j] > 0.f ? acc[j] : 0.f;   // relu
    o += a*Wd2[j];
  }
  out[i] = o;
}

// ---------------------------------------------------------------------------
extern "C" void kernel_launch(void* const* d_in, const int* in_sizes, int n_in,
                              void* d_out, int out_size, void* d_ws, size_t ws_size,
                              hipStream_t stream){
  (void)in_sizes; (void)n_in; (void)out_size; (void)ws_size;
  const float* x_seq  = (const float*)d_in[0];
  const int*   eidx   = (const int*)d_in[1];
  const float* W1     = (const float*)d_in[2];
  const float* a_src1 = (const float*)d_in[3];
  const float* a_dst1 = (const float*)d_in[4];
  const float* b1     = (const float*)d_in[5];
  const float* W2     = (const float*)d_in[6];
  const float* a_src2 = (const float*)d_in[7];
  const float* a_dst2 = (const float*)d_in[8];
  const float* b2     = (const float*)d_in[9];
  const float* Wih0   = (const float*)d_in[10];
  const float* Whh0   = (const float*)d_in[11];
  const float* bih0   = (const float*)d_in[12];
  const float* bhh0   = (const float*)d_in[13];
  const float* Wih1   = (const float*)d_in[14];
  const float* Whh1   = (const float*)d_in[15];
  const float* bih1   = (const float*)d_in[16];
  const float* bhh1   = (const float*)d_in[17];
  const float* Wd1    = (const float*)d_in[18];
  const float* bd1    = (const float*)d_in[19];
  const float* Wd2    = (const float*)d_in[20];
  const float* bd2    = (const float*)d_in[21];

  char* w = (char*)d_ws;
  size_t off = 0;
  auto alloc = [&](size_t bytes)->char*{
    size_t o = off; off = (off + bytes + 255) & ~(size_t)255; return w + o;
  };
  int*       srcA   = (int*)alloc(sizeof(int)*ETOT);
  int*       dstA   = (int*)alloc(sizeof(int)*ETOT);
  _Float16*  x16    = (_Float16*)alloc(sizeof(_Float16)*(size_t)NN*FIN);
  float*     h1     = (float*)alloc(sizeof(float)*(size_t)NN*256);
  float*     out1   = (float*)alloc(sizeof(float)*(size_t)NN*256);
  _Float16*  h1a16  = (_Float16*)alloc(sizeof(_Float16)*(size_t)NN*256);
  float*     alS1   = (float*)alloc(sizeof(float)*(size_t)NN*HEADS);
  float*     alD1   = (float*)alloc(sizeof(float)*(size_t)NN*HEADS);
  float*     m1     = (float*)alloc(sizeof(float)*(size_t)NN*HEADS);
  float*     s1     = (float*)alloc(sizeof(float)*(size_t)NN*HEADS);
  float*     ebuf   = (float*)alloc(sizeof(float)*(size_t)ETOT*HEADS);
  float*     h2     = (float*)alloc(sizeof(float)*(size_t)NN*HIDC);
  float*     out2   = (float*)alloc(sizeof(float)*(size_t)NN*HIDC);
  _Float16*  h2a16  = (_Float16*)alloc(sizeof(_Float16)*(size_t)NN*HIDC);
  float*     alS2   = (float*)alloc(sizeof(float)*(size_t)NN);
  float*     alD2   = (float*)alloc(sizeof(float)*(size_t)NN);
  float*     m2     = (float*)alloc(sizeof(float)*(size_t)NN);
  float*     s2     = (float*)alloc(sizeof(float)*(size_t)NN);
  _Float16*  emb16  = (_Float16*)alloc(sizeof(_Float16)*(size_t)EE*GH);
  float*     h0     = (float*)alloc(sizeof(float)*(size_t)EE*GH);
  _Float16*  h0_16  = (_Float16*)alloc(sizeof(_Float16)*(size_t)EE*GH);
  float*     hs1    = (float*)alloc(sizeof(float)*(size_t)EE*GH);
  _Float16*  hs1_16 = (_Float16*)alloc(sizeof(_Float16)*(size_t)EE*GH);
  _Float16*  imgW1  = (_Float16*)alloc(sizeof(_Float16)*16*1*32*16);
  _Float16*  imgW2  = (_Float16*)alloc(sizeof(_Float16)*4*8*32*16);
  _Float16*  imgIh0 = (_Float16*)alloc(sizeof(_Float16)*GRU_IMG_HALVES);
  _Float16*  imgHh0 = (_Float16*)alloc(sizeof(_Float16)*GRU_IMG_HALVES);
  _Float16*  imgIh1 = (_Float16*)alloc(sizeof(_Float16)*GRU_IMG_HALVES);
  _Float16*  imgHh1 = (_Float16*)alloc(sizeof(_Float16)*GRU_IMG_HALVES);

  const int BLK = 256;
  auto grid = [](long long n, int blk){ return (unsigned)((n + blk - 1)/blk); };

  // allow 192KB dynamic LDS for the GRU kernel (WGP has 320KB)
  static bool attr_set = [](){
    hipFuncSetAttribute((const void*)gru_step,
                        hipFuncAttributeMaxDynamicSharedMemorySize, GRU_LDS_BYTES);
    return true;
  }();
  (void)attr_set;

  // one-time prep (re-done every call; deterministic)
  build_edges<<<grid(ETOT,BLK),BLK,0,stream>>>(eidx, srcA, dstA);
  prep_bimg<<<grid(16*1*32,BLK),BLK,0,stream>>>(W1,   imgW1,  32, 256, 0);
  prep_bimg<<<grid(4*8*32, BLK),BLK,0,stream>>>(W2,   imgW2, 256,  64, 0);
  prep_bimg<<<grid(24*4*32,BLK),BLK,0,stream>>>(Wih0, imgIh0, GH, 384, 1);
  prep_bimg<<<grid(24*4*32,BLK),BLK,0,stream>>>(Whh0, imgHh0, GH, 384, 1);
  prep_bimg<<<grid(24*4*32,BLK),BLK,0,stream>>>(Wih1, imgIh1, GH, 384, 1);
  prep_bimg<<<grid(24*4*32,BLK),BLK,0,stream>>>(Whh1, imgHh1, GH, 384, 1);
  hipMemsetAsync(h0,     0, sizeof(float)*(size_t)EE*GH, stream);
  hipMemsetAsync(h0_16,  0, sizeof(_Float16)*(size_t)EE*GH, stream);
  hipMemsetAsync(hs1,    0, sizeof(float)*(size_t)EE*GH, stream);
  hipMemsetAsync(hs1_16, 0, sizeof(_Float16)*(size_t)EE*GH, stream);

  const int ntiles = NN/16;      // 1250
  const int etiles = EE/16;      // 20000
  const int wavesPerBlk = BLK/32;
  const int gruBlocks = 640;     // grid-stride; LDS fill amortized ~4 tiles/wave

  for (int t = 0; t < T_STEPS; ++t){
    const float* xt = x_seq + (size_t)t*NN*FIN;
    // ---- GAT layer 1 ----
    conv_f16<<<grid((long long)NN*FIN,BLK),BLK,0,stream>>>(xt, x16, NN*FIN);
    wmma_linear<1><<<grid(ntiles,wavesPerBlk),BLK,0,stream>>>(x16, FIN, imgW1, h1, 256, 16, ntiles);
    compute_al<<<grid((long long)NN*HEADS,BLK),BLK,0,stream>>>(h1, a_src1, a_dst1, alS1, alD1, NN, HEADS, HIDC);
    init_ms<<<grid((long long)NN*HEADS,BLK),BLK,0,stream>>>(m1, s1, NN*HEADS);
    hipMemsetAsync(out1, 0, sizeof(float)*(size_t)NN*256, stream);
    attn_edge1<<<grid((long long)ETOT*HEADS,BLK),BLK,0,stream>>>(srcA, dstA, alS1, alD1, ebuf, m1, HEADS);
    attn_edge2<<<grid((long long)ETOT*HEADS,BLK),BLK,0,stream>>>(dstA, ebuf, m1, s1, HEADS);
    attn_edge3<<<grid((long long)ETOT*HEADS,BLK),BLK,0,stream>>>(srcA, dstA, ebuf, s1, h1, out1, HEADS, HIDC);
    finalize_elu<<<grid((long long)NN*256,BLK),BLK,0,stream>>>(out1, b1, h1a16, NN*256, 256);
    // ---- GAT layer 2 ----
    wmma_linear<8><<<grid(ntiles,wavesPerBlk),BLK,0,stream>>>(h1a16, 256, imgW2, h2, HIDC, 4, ntiles);
    compute_al<<<grid((long long)NN,BLK),BLK,0,stream>>>(h2, a_src2, a_dst2, alS2, alD2, NN, 1, HIDC);
    init_ms<<<grid((long long)NN,BLK),BLK,0,stream>>>(m2, s2, NN);
    hipMemsetAsync(out2, 0, sizeof(float)*(size_t)NN*HIDC, stream);
    attn_edge1<<<grid((long long)ETOT,BLK),BLK,0,stream>>>(srcA, dstA, alS2, alD2, ebuf, m2, 1);
    attn_edge2<<<grid((long long)ETOT,BLK),BLK,0,stream>>>(dstA, ebuf, m2, s2, 1);
    attn_edge3<<<grid((long long)ETOT,BLK),BLK,0,stream>>>(srcA, dstA, ebuf, s2, h2, out2, 1, HIDC);
    finalize_add<<<grid((long long)NN*HIDC,BLK),BLK,0,stream>>>(out2, b2, h2a16, NN*HIDC, HIDC);
    // ---- edge embedding + fused GRU steps ----
    gather_emb<<<grid((long long)EE*HIDC,BLK),BLK,0,stream>>>(eidx, h2a16, emb16);
    gru_step<<<gruBlocks,BLK,GRU_LDS_BYTES,stream>>>(emb16, h0_16,  h0,  imgIh0, imgHh0, bih0, bhh0, etiles);
    gru_step<<<gruBlocks,BLK,GRU_LDS_BYTES,stream>>>(h0_16, hs1_16, hs1, imgIh1, imgHh1, bih1, bhh1, etiles);
  }

  // ---- decoder ----
  decoder_k<<<grid((long long)EE,BLK),BLK,0,stream>>>(hs1, Wd1, bd1, Wd2, bd2, (float*)d_out, EE);
}
